// CausalSelfAttention_14310831030712
// MI455X (gfx1250) — compile-verified, matrix operand order checked by reading, not run on hardware
//
#include <hip/hip_runtime.h>

typedef __attribute__((ext_vector_type(16))) __bf16 v16bf;
typedef __attribute__((ext_vector_type(8)))  float  v8f;
typedef __attribute__((__vector_size__(16))) int vi4;

#define BB 2
#define SS 2048
#define DD 1024
#define HH 16
#define HD 64

#if __has_builtin(__builtin_amdgcn_global_load_async_to_lds_b128)
#define HAVE_ASYNC_LDS 1
#else
#define HAVE_ASYNC_LDS 0
#endif

#define AS1 __attribute__((address_space(1)))
#define AS3 __attribute__((address_space(3)))

union FragU { uint4 q[2]; v16bf v; };

__device__ __forceinline__ unsigned short f2bfu(float f) {
  unsigned u = __float_as_uint(f);
  unsigned r = u + 0x7FFFu + ((u >> 16) & 1u);
  return (unsigned short)(r >> 16);
}

// CDNA5 16-bit A/B fragment layout: lane l holds row (l & 15);
// elements 0..7 = K k0+koff..+7, elements 8..15 = K k0+koff+16..+23,
// koff = (l>>4)*8.
__device__ __forceinline__ void ldfrag(FragU& f, const unsigned short* base,
                                       int row, int stride, int k0, int lane) {
  const unsigned short* p = base + (size_t)row * stride + k0 + ((lane >> 4) << 3);
  f.q[0] = *(const uint4*)(p);
  f.q[1] = *(const uint4*)(p + 16);
}

__device__ __forceinline__ v8f wmma_bf16(v16bf a, v16bf b, v8f c) {
  return __builtin_amdgcn_wmma_f32_16x16x32_bf16(false, a, false, b, (short)0, c,
                                                 false, false);
}

__device__ __forceinline__ void async_cp16(const unsigned short* g,
                                           unsigned short* l) {
#if HAVE_ASYNC_LDS
  __builtin_amdgcn_global_load_async_to_lds_b128(
      (AS1 vi4*)(void*)g, (AS3 vi4*)(void*)l, 0, 0);
#else
  *(uint4*)l = *(const uint4*)g;
#endif
}

__device__ __forceinline__ void async_wait0() {
#if HAVE_ASYNC_LDS
#if __has_builtin(__builtin_amdgcn_s_wait_asynccnt)
  __builtin_amdgcn_s_wait_asynccnt(0);
#else
  asm volatile("s_wait_asynccnt 0x0" ::: "memory");
#endif
#endif
}

// ---------------- conversion kernels ----------------
__global__ void k_conv_bf16(const float* __restrict__ in,
                            unsigned short* __restrict__ out, int n) {
  int i = blockIdx.x * blockDim.x + threadIdx.x;
  if (i < n) out[i] = f2bfu(in[i]);
}

// in [rows, cols] fp32 -> out [cols, rows] bf16
__global__ void k_trans_bf16(const float* __restrict__ in,
                             unsigned short* __restrict__ out,
                             int rows, int cols) {
  size_t t = (size_t)blockIdx.x * blockDim.x + threadIdx.x;
  if (t >= (size_t)rows * cols) return;
  int o = (int)(t % cols);
  int i = (int)(t / cols);
  out[(size_t)o * rows + i] = f2bfu(in[t]);
}

// ---- GEMM inner loop: wave = 32x64 tile, base-pointer + immediate offsets.
// acc[2][4], A row-major [*,1024] bf16, B as W^T rows [*,1024] bf16.
__device__ __forceinline__ void gemm_32x64(const unsigned short* __restrict__ A,
                                           const unsigned short* __restrict__ Bm,
                                           int m_base, int n_base, int lane,
                                           v8f acc[2][4]) {
  const int nl = lane & 15;
  const int koffe = (lane >> 4) << 3;
  const unsigned short* pA = A  + (size_t)(m_base + nl) * DD + koffe;
  const unsigned short* pB = Bm + (size_t)(n_base + nl) * DD + koffe;

#pragma unroll
  for (int mt = 0; mt < 2; ++mt)
#pragma unroll
    for (int nt = 0; nt < 4; ++nt)
      acc[mt][nt] = (v8f){0.f,0.f,0.f,0.f,0.f,0.f,0.f,0.f};

#pragma unroll 1
  for (int kk = 0; kk < DD; kk += 32) {
    FragU a0, a1;
    a0.q[0] = *(const uint4*)(pA);
    a0.q[1] = *(const uint4*)(pA + 16);
    a1.q[0] = *(const uint4*)(pA + 16 * DD);
    a1.q[1] = *(const uint4*)(pA + 16 * DD + 16);
#pragma unroll
    for (int nt = 0; nt < 4; ++nt) {
      FragU b;
      b.q[0] = *(const uint4*)(pB + nt * 16 * DD);
      b.q[1] = *(const uint4*)(pB + nt * 16 * DD + 16);
      acc[0][nt] = wmma_bf16(a0.v, b.v, acc[0][nt]);
      acc[1][nt] = wmma_bf16(a1.v, b.v, acc[1][nt]);
    }
    pA += 32;
    pB += 32;
  }
}

// ---------------- QKV GEMM ----------------
__global__ void __launch_bounds__(32)
k_qkv_gemm(const unsigned short* __restrict__ xb,   // [4096,1024]
           const unsigned short* __restrict__ wta,  // [3072,1024]
           const float* __restrict__ bias,          // [3072]
           unsigned short* __restrict__ Qb,         // [B,H,S,HD]
           unsigned short* __restrict__ Kb,         // [B,H,S,HD]
           unsigned short* __restrict__ Vtb) {      // [B,H,HD,S]
  const int lane = threadIdx.x;
  const int nl = lane & 15, hi = lane >> 4;
  const int m_base = blockIdx.x * 32;
  const int n_base = blockIdx.y * 64;

  v8f acc[2][4];
  gemm_32x64(xb, wta, m_base, n_base, lane, acc);

  const int region = blockIdx.y / 16;     // 0=Q 1=K 2=V
  const int h = blockIdx.y % 16;
#pragma unroll
  for (int mt = 0; mt < 2; ++mt) {
#pragma unroll
    for (int nt = 0; nt < 4; ++nt) {
#pragma unroll
      for (int r = 0; r < 8; ++r) {
        const int m  = m_base + mt * 16 + r + 8 * hi;
        const int bg = m / SS, s = m % SS;
        const int hd = nt * 16 + nl;
        const int og = n_base + nt * 16 + nl;
        float v = acc[mt][nt][r] + bias[og];
        if (region == 0) {
          Qb[(((size_t)bg * HH + h) * SS + s) * HD + hd] = f2bfu(v * 0.125f);
        } else if (region == 1) {
          Kb[(((size_t)bg * HH + h) * SS + s) * HD + hd] = f2bfu(v);
        } else {
          Vtb[(((size_t)bg * HH + h) * HD + hd) * SS + s] = f2bfu(v);
        }
      }
    }
  }
}

// ---------------- flash attention: 4 waves/block, LDS-staged K/V ----------
__global__ void __launch_bounds__(128)
k_flash(const unsigned short* __restrict__ Qb,
        const unsigned short* __restrict__ Kb,
        const unsigned short* __restrict__ Vtb,
        unsigned short* __restrict__ Yb) {  // [B*S, D] bf16
  __shared__ __align__(16) unsigned short Kt[32 * 64];   // [key][hd]   4 KB
  __shared__ __align__(16) unsigned short Vt[64 * 32];   // [hd][key]   4 KB
  __shared__ __align__(16) unsigned short Pl[4][16 * 32];  // per-wave P tiles

  const int tid = threadIdx.x;
  const int lane = tid & 31;
  const int w = tid >> 5;                 // wave id 0..3
  const int nl = lane & 15, hi = lane >> 4, koff = hi * 8;
  const int qb0 = blockIdx.x * 64;        // block q range [qb0, qb0+64)
  const int q_base = qb0 + w * 16;        // this wave's q tile
  const int h = blockIdx.y, bg = blockIdx.z;
  const size_t bh = (size_t)bg * HH + h;

  const unsigned short* Qh = Qb  + bh * SS * HD;
  const unsigned short* Kh = Kb  + bh * SS * HD;
  const unsigned short* Vh = Vtb + bh * HD * SS;

  FragU qa0, qa1;
  ldfrag(qa0, Qh, q_base + nl, HD, 0,  lane);
  ldfrag(qa1, Qh, q_base + nl, HD, 32, lane);

  float rmax[8], rsum[8];
  v8f o[4];
#pragma unroll
  for (int r = 0; r < 8; ++r) { rmax[r] = -1e30f; rsum[r] = 0.f; }
#pragma unroll
  for (int t = 0; t < 4; ++t) o[t] = (v8f){0.f,0.f,0.f,0.f,0.f,0.f,0.f,0.f};

  const v8f z = (v8f){0.f,0.f,0.f,0.f,0.f,0.f,0.f,0.f};
  unsigned short* Pw = &Pl[w][0];

#pragma unroll 1
  for (int kb = 0; kb <= qb0 + 63; kb += 32) {
    __syncthreads();  // protect LDS WAR vs previous iteration's readers

    // cooperatively stage K chunk [kb..kb+32)x64 and Vt chunk 64x[kb..kb+32)
#pragma unroll
    for (int i = 0; i < 2; ++i) {
      const int c = tid + 128 * i;        // 0..255 sixteen-byte chunks
      async_cp16(Kh + (size_t)kb * HD + c * 8, &Kt[c * 8]);
      const int hd = c >> 2, part = c & 3;
      async_cp16(Vh + (size_t)hd * SS + kb + part * 8,
                 &Vt[hd * 32 + part * 8]);
    }
    async_wait0();
    __syncthreads();

    if (kb <= q_base + 15) {   // wave-uniform: EXEC stays all-ones for WMMA
      // scores from LDS K tile
      v8f s0 = z, s1 = z;
      FragU kf;
      ldfrag(kf, Kt, nl,      HD, 0,  lane); s0 = wmma_bf16(qa0.v, kf.v, s0);
      ldfrag(kf, Kt, nl,      HD, 32, lane); s0 = wmma_bf16(qa1.v, kf.v, s0);
      ldfrag(kf, Kt, 16 + nl, HD, 0,  lane); s1 = wmma_bf16(qa0.v, kf.v, s1);
      ldfrag(kf, Kt, 16 + nl, HD, 32, lane); s1 = wmma_bf16(qa1.v, kf.v, s1);

      // causal mask (C layout: element (row r+8*hi, col nl))
      if (kb + 31 > q_base) {
#pragma unroll
        for (int r = 0; r < 8; ++r) {
          const int m = q_base + r + 8 * hi;
          if (kb + nl      > m) s0[r] = -1e30f;
          if (kb + 16 + nl > m) s1[r] = -1e30f;
        }
      }

      // online softmax stats
      float alpha[8];
#pragma unroll
      for (int r = 0; r < 8; ++r) {
        float cm = fmaxf(s0[r], s1[r]);
        for (int mk = 1; mk <= 8; mk <<= 1) cm = fmaxf(cm, __shfl_xor(cm, mk, 32));
        const float nm = fmaxf(rmax[r], cm);
        alpha[r] = __expf(rmax[r] - nm);
        rmax[r] = nm;
      }

      // p = exp(s - rowmax); bf16 P tile to per-wave LDS region
#pragma unroll
      for (int r = 0; r < 8; ++r) {
        const float p0 = __expf(s0[r] - rmax[r]);
        const float p1 = __expf(s1[r] - rmax[r]);
        const int m = r + 8 * hi;
        Pw[m * 32 + nl]      = f2bfu(p0);
        Pw[m * 32 + 16 + nl] = f2bfu(p1);
        float ps = p0 + p1;
        for (int mk = 1; mk <= 8; mk <<= 1) ps += __shfl_xor(ps, mk, 32);
        rsum[r] = rsum[r] * alpha[r] + ps;
      }
      // same-wave DS ops are in-order (ISA ch.11): no barrier needed here

      FragU pa;
      {
        const unsigned short* pp = Pw + (size_t)nl * 32 + koff;
        pa.q[0] = *(const uint4*)(pp);
        pa.q[1] = *(const uint4*)(pp + 16);
      }
#pragma unroll
      for (int t = 0; t < 4; ++t) {
#pragma unroll
        for (int r = 0; r < 8; ++r) o[t][r] *= alpha[r];
        FragU vf; ldfrag(vf, Vt, t * 16 + nl, 32, 0, lane);
        o[t] = wmma_bf16(pa.v, vf.v, o[t]);
      }
    }
  }

  // normalize + store bf16 y tile
#pragma unroll
  for (int t = 0; t < 4; ++t) {
#pragma unroll
    for (int r = 0; r < 8; ++r) {
      const int m_row = bg * SS + q_base + r + 8 * hi;
      const int col = h * HD + t * 16 + nl;
      Yb[(size_t)m_row * DD + col] = f2bfu(o[t][r] / rsum[r]);
    }
  }
}

// ---------------- output projection GEMM ----------------
__global__ void __launch_bounds__(32)
k_proj_gemm(const unsigned short* __restrict__ yb,   // [4096,1024]
            const unsigned short* __restrict__ wtp,  // [1024,1024]
            const float* __restrict__ bias,          // [1024]
            float* __restrict__ out) {               // [4096,1024]
  const int lane = threadIdx.x;
  const int nl = lane & 15, hi = lane >> 4;
  const int m_base = blockIdx.x * 32;
  const int n_base = blockIdx.y * 64;

  v8f acc[2][4];
  gemm_32x64(yb, wtp, m_base, n_base, lane, acc);

#pragma unroll
  for (int mt = 0; mt < 2; ++mt) {
#pragma unroll
    for (int nt = 0; nt < 4; ++nt) {
#pragma unroll
      for (int r = 0; r < 8; ++r) {
        const int m = m_base + mt * 16 + r + 8 * hi;
        const int n = n_base + nt * 16 + nl;
        out[(size_t)m * DD + n] = acc[mt][nt][r] + bias[n];
      }
    }
  }
}

extern "C" void kernel_launch(void* const* d_in, const int* in_sizes, int n_in,
                              void* d_out, int out_size, void* d_ws, size_t ws_size,
                              hipStream_t stream) {
  const float* x      = (const float*)d_in[0];
  const float* W_attn = (const float*)d_in[1];
  const float* b_attn = (const float*)d_in[2];
  const float* W_proj = (const float*)d_in[3];
  const float* b_proj = (const float*)d_in[4];
  float* out = (float*)d_out;

  char* ws = (char*)d_ws;
  unsigned short* xb  = (unsigned short*)(ws);                            // 8 MiB
  unsigned short* Wta = (unsigned short*)(ws + ( 8ull << 20));            // 6 MiB
  unsigned short* Wtp = (unsigned short*)(ws + (14ull << 20));            // 2 MiB
  unsigned short* Qb  = (unsigned short*)(ws + (16ull << 20));            // 8 MiB
  unsigned short* Kb  = (unsigned short*)(ws + (24ull << 20));            // 8 MiB
  unsigned short* Vtb = (unsigned short*)(ws + (32ull << 20));            // 8 MiB
  unsigned short* Yb  = (unsigned short*)(ws + (40ull << 20));            // 8 MiB

  const int n_x = BB * SS * DD;          // 4194304
  const int n_wa = DD * 3 * DD;          // 3145728
  const int n_wp = DD * DD;              // 1048576

  k_conv_bf16<<<(n_x + 255) / 256, 256, 0, stream>>>(x, xb, n_x);
  k_trans_bf16<<<(n_wa + 255) / 256, 256, 0, stream>>>(W_attn, Wta, DD, 3 * DD);
  k_trans_bf16<<<(n_wp + 255) / 256, 256, 0, stream>>>(W_proj, Wtp, DD, DD);

  k_qkv_gemm<<<dim3(BB * SS / 32, 3 * DD / 64), 32, 0, stream>>>(
      xb, Wta, b_attn, Qb, Kb, Vtb);

  k_flash<<<dim3(SS / 64, HH, BB), 128, 0, stream>>>(Qb, Kb, Vtb, Yb);

  k_proj_gemm<<<dim3(BB * SS / 32, DD / 64), 32, 0, stream>>>(
      Yb, Wtp, b_proj, out);
}
